// SparseLinear_6820408066325
// MI455X (gfx1250) — matrix operand order repaired and checked by reading, not compiled, not run on hardware
//
#include <hip/hip_runtime.h>

// ---------------------------------------------------------------------------
// SparseLinear on MI455X (gfx1250):
//   1) memset dense W (fp32) in workspace
//   2) COO scatter-add (duplicates summed) via global_atomic_add_f32
//   3) pre-convert: X -> bf16, dense W -> bf16 transposed (K-major)
//   4) bf16 WMMA GEMM, double-buffered async global->LDS staging
// Fallback (small workspace): round-1 on-the-fly-conversion WMMA GEMM.
// ---------------------------------------------------------------------------

#define IN_F  4096
#define OUT_F 4096
#define NTOK  8192

typedef __attribute__((ext_vector_type(16))) __bf16 v16bf;
typedef __attribute__((ext_vector_type(8)))  __bf16 v8bf;
typedef __attribute__((ext_vector_type(8)))  float  v8f;
typedef __attribute__((ext_vector_type(4)))  int    v4i;

#define AS1 __attribute__((address_space(1)))
#define AS3 __attribute__((address_space(3)))

#if __has_builtin(__builtin_amdgcn_global_load_async_to_lds_b128)
#define HAVE_ASYNC_LDS 1
#endif

__device__ __forceinline__ unsigned short f32_to_bf16_rne(float f) {
    unsigned int u = __float_as_uint(f);
    u += 0x7FFFu + ((u >> 16) & 1u);   // round-to-nearest-even
    return (unsigned short)(u >> 16);
}

__device__ __forceinline__ void wait_async_all() {
#if defined(HAVE_ASYNC_LDS)
#if __has_builtin(__builtin_amdgcn_s_wait_asynccnt)
    __builtin_amdgcn_s_wait_asynccnt(0);
#else
    asm volatile("s_wait_asynccnt 0" ::: "memory");
#endif
#endif
}

// --------------------------- COO scatter-add -------------------------------
__global__ __launch_bounds__(256) void coo_scatter_kernel(
    const float* __restrict__ w,
    const int*   __restrict__ rows,
    const int*   __restrict__ cols,
    float*       __restrict__ Wd,
    int nnz)
{
    int i      = blockIdx.x * blockDim.x + threadIdx.x;
    int stride = gridDim.x * blockDim.x;
    for (; i < nnz; i += stride) {
        atomicAdd(&Wd[rows[i] * IN_F + cols[i]], w[i]);
    }
}

// --------------------- X fp32 -> bf16 (straight convert) -------------------
__global__ __launch_bounds__(256) void cvt_x_bf16_kernel(
    const float* __restrict__ X, unsigned short* __restrict__ Xh, int n4)
{
    int i      = blockIdx.x * blockDim.x + threadIdx.x;
    int stride = gridDim.x * blockDim.x;
    for (; i < n4; i += stride) {
        float4 v = ((const float4*)X)[i];
        ushort4 o;
        o.x = f32_to_bf16_rne(v.x);
        o.y = f32_to_bf16_rne(v.y);
        o.z = f32_to_bf16_rne(v.z);
        o.w = f32_to_bf16_rne(v.w);
        ((ushort4*)Xh)[i] = o;
    }
}

// ------------- W fp32 (o,i) -> bf16 transposed (i,o), LDS-tiled ------------
__global__ __launch_bounds__(256) void cvt_wT_bf16_kernel(
    const float* __restrict__ Wd, unsigned short* __restrict__ WhT)
{
    __shared__ float Tf[32][33];               // +1 pad: no bank conflicts
    const int go0 = blockIdx.x * 32;           // o tile base
    const int gi0 = blockIdx.y * 32;           // i tile base
    const int tid = threadIdx.x;
    #pragma unroll
    for (int q = 0; q < 4; ++q) {
        int idx = tid + q * 256;               // 0..1023
        int r = idx >> 5, c = idx & 31;        // r = o-local, c = i-local
        Tf[r][c] = Wd[(size_t)(go0 + r) * IN_F + gi0 + c];
    }
    __syncthreads();
    {
        int r  = tid >> 3;                     // i-local 0..31
        int c4 = (tid & 7) << 2;               // o-local 0,4,...,28
        ushort4 o;
        o.x = f32_to_bf16_rne(Tf[c4 + 0][r]);
        o.y = f32_to_bf16_rne(Tf[c4 + 1][r]);
        o.z = f32_to_bf16_rne(Tf[c4 + 2][r]);
        o.w = f32_to_bf16_rne(Tf[c4 + 3][r]);
        *(ushort4*)&WhT[(size_t)(gi0 + r) * OUT_F + go0 + c4] = o;
    }
}

// ===================== primary GEMM: bf16 in, async staging =================
// Block tile 128M x 128N, BK=32, 256 threads = 8 waves (4M x 2N),
// wave tile 32x64 -> 8 x v_wmma_f32_16x16x32_bf16 per K-step.
#define BM 128
#define BN 128
#define BK 32

__global__ __launch_bounds__(256) void sparse_linear_wmma_async_kernel(
    const unsigned short* __restrict__ Xh,   // NTOK x IN_F   bf16 (m-major)
    const unsigned short* __restrict__ WhT,  // IN_F x OUT_F  bf16 (k-major!)
    const float* __restrict__ bias,
    float*       __restrict__ Out)
{
    __shared__ __align__(32) unsigned short Xs[2][BM][BK];  // [m][k] 2x8KB
    __shared__ __align__(32) unsigned short Wt[2][BK][BN];  // [k][n] 2x8KB

    const int tid  = threadIdx.x;
    const int lane = tid & 31;
    const int wave = tid >> 5;
    const int wm   = (wave & 3) * 32;
    const int wn   = (wave >> 2) * 64;
    const int gn0  = blockIdx.x * BN;
    const int gm0  = blockIdx.y * BM;

    // stage one K-slab (pure bf16 copies; async DMA to LDS when available)
    auto stage = [&](int buf, int k0) {
        #pragma unroll
        for (int q = 0; q < 2; ++q) {
            int c = tid + q * 256;                       // 0..511 chunk id
            // X chunk: 4 x 16B per 32-half row
            int xr = c >> 2, xc = (c & 3) << 3;
            const unsigned short* gx = &Xh[(size_t)(gm0 + xr) * IN_F + k0 + xc];
            unsigned short*       lx = &Xs[buf][xr][xc];
            // W chunk: 16 x 16B per 128-half K-row
            int wk = c >> 4, wc = (c & 15) << 3;
            const unsigned short* gw = &WhT[(size_t)(k0 + wk) * OUT_F + gn0 + wc];
            unsigned short*       lw = &Wt[buf][wk][wc];
#if defined(HAVE_ASYNC_LDS)
            __builtin_amdgcn_global_load_async_to_lds_b128(
                (AS1 v4i*)(AS1 void*)gx, (AS3 v4i*)(AS3 void*)lx, 0, 0);
            __builtin_amdgcn_global_load_async_to_lds_b128(
                (AS1 v4i*)(AS1 void*)gw, (AS3 v4i*)(AS3 void*)lw, 0, 0);
#else
            *(float4*)lx = *(const float4*)gx;
            *(float4*)lw = *(const float4*)gw;
#endif
        }
    };

    v8f acc[2][4] = {};   // [m subtile][n subtile]

    stage(0, 0);
    wait_async_all();
    __syncthreads();

    const int kb = (lane >> 4) << 3;   // A frag K base: 0 or 8 halves
    const int ma = lane & 15;

    for (int t = 0; t < IN_F / BK; ++t) {
        const int buf = t & 1;
        if (t + 1 < IN_F / BK) stage(buf ^ 1, (t + 1) * BK);

        // A fragments (ISA 16-bit A 16x32 layout)
        v16bf a[2];
        #pragma unroll
        for (int mi = 0; mi < 2; ++mi) {
            v8bf lo = *(const v8bf*)&Xs[buf][wm + mi * 16 + ma][kb];
            v8bf hi = *(const v8bf*)&Xs[buf][wm + mi * 16 + ma][kb + 16];
            a[mi] = __builtin_shufflevector(lo, hi,
                     0,1,2,3,4,5,6,7,8,9,10,11,12,13,14,15);
        }
        // B fragments: lane l holds column l%16, K in the same table order as A
        #pragma unroll
        for (int ni = 0; ni < 4; ++ni) {
            v16bf b;
            #pragma unroll
            for (int s = 0; s < 16; ++s)
                b[s] = __builtin_bit_cast(__bf16, Wt[buf][kb + (s & 7) + ((s >> 3) << 4)][wn + ni * 16 + ma]);
            #pragma unroll
            for (int mi = 0; mi < 2; ++mi) {
                acc[mi][ni] = __builtin_amdgcn_wmma_f32_16x16x32_bf16(
                    false, a[mi], false, b, (short)0, acc[mi][ni], false, false);
            }
        }
        wait_async_all();   // next slab landed before we flip buffers
        __syncthreads();    // all waves done reading buf before it is refilled
    }

    // epilogue: C/D layout: col = lane&15, VGPR r -> row r (+8 for lane>=16)
    const int nc    = lane & 15;
    const int rbase = (lane >> 4) << 3;
    #pragma unroll
    for (int mi = 0; mi < 2; ++mi) {
        #pragma unroll
        for (int ni = 0; ni < 4; ++ni) {
            const int gn = gn0 + wn + ni * 16 + nc;
            const float bv = bias[gn];
            const int gm = gm0 + wm + mi * 16 + rbase;
            #pragma unroll
            for (int r = 0; r < 8; ++r) {
                Out[(size_t)(gm + r) * OUT_F + gn] = acc[mi][ni][r] + bv;
            }
        }
    }
}

// ============== fallback GEMM (round-1: fp32 W, on-the-fly cvt) =============
#define FBM 128
#define FBN 64
#define FBK 32

__global__ __launch_bounds__(256) void sparse_linear_wmma_small_ws_kernel(
    const float* __restrict__ X,
    const float* __restrict__ Wd,
    const float* __restrict__ bias,
    float*       __restrict__ Out)
{
    __shared__ __align__(32) unsigned short Xs[FBM][FBK];
    __shared__ __align__(32) unsigned short Wt[FBK][FBN];

    const int tid  = threadIdx.x;
    const int lane = tid & 31;
    const int wave = tid >> 5;
    const int wm   = (wave & 3) * 32;
    const int wn   = (wave >> 2) * 32;
    const int gn0  = blockIdx.x * FBN;
    const int gm0  = blockIdx.y * FBM;

    v8f c00 = {}, c01 = {}, c10 = {}, c11 = {};

    for (int k0 = 0; k0 < IN_F; k0 += FBK) {
        #pragma unroll
        for (int q = 0; q < 4; ++q) {
            int idx = tid + q * 256;
            int r = idx >> 3, c4 = (idx & 7) << 2;
            const float4 v = *(const float4*)&X[(size_t)(gm0 + r) * IN_F + k0 + c4];
            unsigned short* dst = &Xs[r][c4];
            dst[0] = f32_to_bf16_rne(v.x);
            dst[1] = f32_to_bf16_rne(v.y);
            dst[2] = f32_to_bf16_rne(v.z);
            dst[3] = f32_to_bf16_rne(v.w);
        }
        #pragma unroll
        for (int q = 0; q < 2; ++q) {
            int idx = tid + q * 256;
            int r = idx >> 3, c4 = (idx & 7) << 2;
            const float4 v = *(const float4*)&Wd[(size_t)(gn0 + r) * IN_F + k0 + c4];
            Wt[c4 + 0][r] = f32_to_bf16_rne(v.x);
            Wt[c4 + 1][r] = f32_to_bf16_rne(v.y);
            Wt[c4 + 2][r] = f32_to_bf16_rne(v.z);
            Wt[c4 + 3][r] = f32_to_bf16_rne(v.w);
        }
        if (k0 + FBK < IN_F) {
            __builtin_prefetch(&X[(size_t)(gm0 + (tid >> 1)) * IN_F + k0 + FBK], 0, 1);
            __builtin_prefetch(&Wd[(size_t)(gn0 + (tid & 63)) * IN_F + k0 + FBK], 0, 1);
        }
        __syncthreads();

        const int kb = (lane >> 4) << 3;
        const int ma = lane & 15;
        v8bf a0lo = *(const v8bf*)&Xs[wm + ma][kb];
        v8bf a0hi = *(const v8bf*)&Xs[wm + ma][kb + 16];
        v8bf a1lo = *(const v8bf*)&Xs[wm + 16 + ma][kb];
        v8bf a1hi = *(const v8bf*)&Xs[wm + 16 + ma][kb + 16];
        v16bf a0 = __builtin_shufflevector(a0lo, a0hi,
                    0,1,2,3,4,5,6,7,8,9,10,11,12,13,14,15);
        v16bf a1 = __builtin_shufflevector(a1lo, a1hi,
                    0,1,2,3,4,5,6,7,8,9,10,11,12,13,14,15);
        v16bf b0, b1;
        #pragma unroll
        for (int s = 0; s < 16; ++s) {
            b0[s] = __builtin_bit_cast(__bf16, Wt[kb + (s & 7) + ((s >> 3) << 4)][wn + ma]);
            b1[s] = __builtin_bit_cast(__bf16, Wt[kb + (s & 7) + ((s >> 3) << 4)][wn + 16 + ma]);
        }

        c00 = __builtin_amdgcn_wmma_f32_16x16x32_bf16(false, a0, false, b0, (short)0, c00, false, false);
        c01 = __builtin_amdgcn_wmma_f32_16x16x32_bf16(false, a0, false, b1, (short)0, c01, false, false);
        c10 = __builtin_amdgcn_wmma_f32_16x16x32_bf16(false, a1, false, b0, (short)0, c10, false, false);
        c11 = __builtin_amdgcn_wmma_f32_16x16x32_bf16(false, a1, false, b1, (short)0, c11, false, false);
        __syncthreads();
    }

    const int nc    = lane & 15;
    const int rbase = (lane >> 4) << 3;
    auto store_tile = [&](const v8f& c, int ms, int ns) {
        const int gn = gn0 + wn + ns + nc;
        const float bv = bias[gn];
        const int gm = gm0 + wm + ms + rbase;
        #pragma unroll
        for (int r = 0; r < 8; ++r)
            Out[(size_t)(gm + r) * OUT_F + gn] = c[r] + bv;
    };
    store_tile(c00, 0, 0);
    store_tile(c01, 0, 16);
    store_tile(c10, 16, 0);
    store_tile(c11, 16, 16);
}

// ------------------------------ launcher -----------------------------------
extern "C" void kernel_launch(void* const* d_in, const int* in_sizes, int n_in,
                              void* d_out, int out_size, void* d_ws, size_t ws_size,
                              hipStream_t stream) {
    const float* X    = (const float*)d_in[0];   // 8192 x 4096 fp32
    const float* w    = (const float*)d_in[1];   // NNZ fp32
    const float* bias = (const float*)d_in[2];   // 4096 fp32
    const int*   idx  = (const int*)d_in[3];     // 2 x NNZ (rows then cols)
    const int    nnz  = in_sizes[1];
    float*       Out  = (float*)d_out;

    const size_t szWd  = (size_t)OUT_F * IN_F * sizeof(float);          // 64 MB
    const size_t szWhT = (size_t)IN_F * OUT_F * sizeof(unsigned short); // 32 MB
    const size_t szXh  = (size_t)NTOK * IN_F * sizeof(unsigned short);  // 64 MB

    float* Wd = (float*)d_ws;

    // 1) zero dense fp32 W, 2) scatter-add COO (both paths need this)
    (void)hipMemsetAsync(Wd, 0, szWd, stream);
    coo_scatter_kernel<<<2048, 256, 0, stream>>>(w, idx, idx + nnz, Wd, nnz);

    if (ws_size >= szWd + szWhT + szXh) {
        unsigned short* WhT = (unsigned short*)((char*)d_ws + szWd);
        unsigned short* Xh  = (unsigned short*)((char*)d_ws + szWd + szWhT);

        // 3) pre-convert: W -> bf16 transposed (K-major), X -> bf16
        cvt_wT_bf16_kernel<<<dim3(OUT_F / 32, IN_F / 32), 256, 0, stream>>>(Wd, WhT);
        cvt_x_bf16_kernel<<<4096, 256, 0, stream>>>(X, (unsigned short*)Xh,
                                                    (NTOK * IN_F) / 4);

        // 4) async double-buffered WMMA GEMM with fused bias
        dim3 grid(OUT_F / BN, NTOK / BM);
        sparse_linear_wmma_async_kernel<<<grid, 256, 0, stream>>>(Xh, WhT, bias, Out);
    } else {
        // fallback: only needs the 64 MB dense fp32 W
        dim3 grid(OUT_F / FBN, NTOK / FBM);
        sparse_linear_wmma_small_ws_kernel<<<grid, 256, 0, stream>>>(X, Wd, bias, Out);
    }
}
